// TGCNCell_27986006901453
// MI455X (gfx1250) — matrix operand order repaired and checked by reading, not compiled, not run on hardware
//
#include <hip/hip_runtime.h>
#include <math.h>

// ---------------------------------------------------------------------------
// CDNA5 (gfx1250) implementation of the hierarchical TGCN cell.
// Heavy lifting (adj @ features, treated dense as in the reference) is done
// with v_wmma_f32_16x16x32_bf16, wave32. Column dim padded 1040 -> 1088 so
// each wave owns exactly 1 M-tile x 4 N-tiles with no tail branches.
// ---------------------------------------------------------------------------

typedef __attribute__((ext_vector_type(16))) __bf16 v16bf;
typedef __attribute__((ext_vector_type(8)))  float  v8f;

namespace {
constexpr int Bn   = 16;     // batch
constexpr int Nf   = 5000;   // fine nodes
constexpr int NCd  = 500;    // coarse nodes
constexpr int NSd  = 50;     // super nodes
constexpr int Ud   = 64;     // hidden units
constexpr int INSZ = 65;     // 1 + U
constexpr int CCOL  = Bn * INSZ;  // 1040 logical GEMM columns
constexpr int CCOLP = 1088;       // padded columns: 68 tiles = 17 groups of 4
constexpr int NGRP  = 17;         // column-tile groups per M-tile

// padded GEMM dims (M multiple of 16, K multiple of 32)
constexpr int MPF = 5008, KPF = 5024;   // fine
constexpr int MPC = 512,  KPC = 512;    // coarse
constexpr int MPS = 64,   KPS = 64;     // super
}

// --------------------------- WMMA GEMM -------------------------------------
// Y[Mp x CCOLP] = A[Mp x Kp](bf16 row-major) @ B, with B given transposed:
// Bt[col][k] (CCOLP rows of length Kp, bf16). Each wave: 1 M-tile x 4 N-tiles,
// branch-free inner loop (EXEC all-ones throughout, as WMMA requires).
__global__ void wmma_gemm(const __bf16* __restrict__ A,
                          const __bf16* __restrict__ Bt,
                          float* __restrict__ Y,
                          int Mp, int Kp) {
  const int lane = threadIdx.x & 31;
  const int wv   = __builtin_amdgcn_readfirstlane(threadIdx.x >> 5);
  const int wave = blockIdx.x * 8 + wv;          // 256 threads = 8 waves
  const int tilesM = Mp >> 4;
  const int mtile  = wave / NGRP;
  const int ng     = wave % NGRP;
  if (mtile >= tilesM) return;                   // scalar (wave-uniform) exit

  const int l15 = lane & 15;
  const __bf16* aRow = A + (size_t)(mtile * 16 + l15) * Kp;
  const int aK = (lane < 16) ? 0 : 16;  // A frag: contiguous 16 K per half-wave
  const int bK = (lane < 16) ? 0 : 16;  // B frag: contiguous 16 K per half-wave

  const __bf16* bRow0 = Bt + (size_t)((ng * 4 + 0) * 16 + l15) * Kp + bK;
  const __bf16* bRow1 = Bt + (size_t)((ng * 4 + 1) * 16 + l15) * Kp + bK;
  const __bf16* bRow2 = Bt + (size_t)((ng * 4 + 2) * 16 + l15) * Kp + bK;
  const __bf16* bRow3 = Bt + (size_t)((ng * 4 + 3) * 16 + l15) * Kp + bK;

  v8f acc0, acc1, acc2, acc3;
#pragma unroll
  for (int v = 0; v < 8; ++v) { acc0[v] = 0.f; acc1[v] = 0.f; acc2[v] = 0.f; acc3[v] = 0.f; }

  const int ksteps = Kp >> 5;
#pragma unroll 2
  for (int kk = 0; kk < ksteps; ++kk) {
    union { v16bf v; __bf16 e[16]; } af;
    const __bf16* ap = aRow + kk * 32 + aK;
#pragma unroll
    for (int j = 0; j < 8; ++j) { af.e[j] = ap[j]; af.e[8 + j] = ap[8 + j]; }

    union { v16bf v; __bf16 e[16]; } b0, b1, b2, b3;
    const int ko = kk * 32;
#pragma unroll
    for (int j = 0; j < 16; ++j) {
      b0.e[j] = bRow0[ko + j];
      b1.e[j] = bRow1[ko + j];
      b2.e[j] = bRow2[ko + j];
      b3.e[j] = bRow3[ko + j];
    }
    acc0 = __builtin_amdgcn_wmma_f32_16x16x32_bf16(false, af.v, false, b0.v, (short)0, acc0, false, false);
    acc1 = __builtin_amdgcn_wmma_f32_16x16x32_bf16(false, af.v, false, b1.v, (short)0, acc1, false, false);
    acc2 = __builtin_amdgcn_wmma_f32_16x16x32_bf16(false, af.v, false, b2.v, (short)0, acc2, false, false);
    acc3 = __builtin_amdgcn_wmma_f32_16x16x32_bf16(false, af.v, false, b3.v, (short)0, acc3, false, false);
  }

  const int mbase = mtile * 16 + ((lane >= 16) ? 8 : 0);
  float* y0 = Y + (size_t)mbase * CCOLP + (ng * 4 + 0) * 16 + l15;
#pragma unroll
  for (int v = 0; v < 8; ++v) {
    float* yr = y0 + (size_t)v * CCOLP;
    yr[0]  = acc0[v];
    yr[16] = acc1[v];
    yr[32] = acc2[v];
    yr[48] = acc3[v];
  }
}

// ------------------------- pack / pointwise kernels -------------------------

// dense fp32 MxK -> bf16 MpxKp (zero padded)
__global__ void pack_mat_bf16(const float* __restrict__ src, __bf16* __restrict__ dst,
                              int M, int K, int Mp, int Kp) {
  size_t idx = (size_t)blockIdx.x * blockDim.x + threadIdx.x;
  size_t tot = (size_t)Mp * Kp;
  if (idx >= tot) return;
  int k = (int)(idx % Kp), m = (int)(idx / Kp);
  float v = (m < M && k < K) ? src[(size_t)m * K + k] : 0.f;
  dst[idx] = (__bf16)v;
}

// Xt[col = b*65+f][k = node]  (CCOLP rows; rows >= 1040 and k >= Nn are zero)
__global__ void pack_x0(__bf16* __restrict__ Xt, const float* __restrict__ xi,
                        const float* __restrict__ h, int Nn, int Kp) {
  size_t idx = (size_t)blockIdx.x * blockDim.x + threadIdx.x;
  size_t tot = (size_t)CCOLP * Kp;
  if (idx >= tot) return;
  int k = (int)(idx % Kp), c = (int)(idx / Kp);
  float v = 0.f;
  if (c < CCOL && k < Nn) {
    int b = c / INSZ, f = c % INSZ;
    v = (f == 0) ? xi[(size_t)b * Nn + k]
                 : h[((size_t)b * Nn + k) * Ud + (f - 1)];
  }
  Xt[idx] = (__bf16)v;
}

// gate: g = Y_row(65) @ W(65x128) + b ; sigmoid ; r -> Xt cols 1..64 (= r*h),
// u -> u_out. Xt col f=0 (xi) stays valid from pack_x0.
__global__ void gate_kernel(const float* __restrict__ Y, const float* __restrict__ h,
                            const float* __restrict__ W, const float* __restrict__ bias,
                            float* __restrict__ u_out, __bf16* __restrict__ Xt,
                            int Nn, int Kp) {
  size_t idx = (size_t)blockIdx.x * blockDim.x + threadIdx.x;
  size_t tot = (size_t)Bn * Nn * 128;
  if (idx >= tot) return;
  int o = (int)(idx & 127);
  size_t rest = idx >> 7;
  int n = (int)(rest % Nn), b = (int)(rest / Nn);
  const float* yp = Y + (size_t)n * CCOLP + b * INSZ;
  float g = bias[o];
#pragma unroll 5
  for (int f = 0; f < INSZ; ++f) g += yp[f] * W[f * 128 + o];
  float sig = 1.f / (1.f + expf(-g));
  if (o < Ud) {
    float hv = h[((size_t)b * Nn + n) * Ud + o];
    Xt[(size_t)(b * INSZ + 1 + o) * Kp + n] = (__bf16)(sig * hv);
  } else {
    u_out[((size_t)b * Nn + n) * Ud + (o - Ud)] = sig;
  }
}

// candidate: c = tanh(Y_row(65) @ W(65x64) + b); out = u*h + (1-u)*c
__global__ void cand_kernel(const float* __restrict__ Y, const float* __restrict__ h,
                            const float* __restrict__ W, const float* __restrict__ bias,
                            const float* __restrict__ u_ws, float* __restrict__ outp,
                            int Nn) {
  size_t idx = (size_t)blockIdx.x * blockDim.x + threadIdx.x;
  size_t tot = (size_t)Bn * Nn * Ud;
  if (idx >= tot) return;
  int j = (int)(idx & 63);
  size_t rest = idx >> 6;
  int n = (int)(rest % Nn), b = (int)(rest / Nn);
  const float* yp = Y + (size_t)n * CCOLP + b * INSZ;
  float c = bias[j];
#pragma unroll 5
  for (int f = 0; f < INSZ; ++f) c += yp[f] * W[f * Ud + j];
  c = tanhf(c);
  size_t off = ((size_t)b * Nn + n) * Ud + j;
  float u = u_ws[off], hv = h[off];
  outp[off] = u * hv + (1.f - u) * c;
}

// xc[b,c] = sum_n afc[c,n] * inputs[b,n]
__global__ void xc_kernel(const float* __restrict__ afc, const float* __restrict__ xi,
                          float* __restrict__ xc) {
  int idx = blockIdx.x * blockDim.x + threadIdx.x;
  if (idx >= Bn * NCd) return;
  int c = idx % NCd, b = idx / NCd;
  const float* ar = afc + (size_t)c * Nf;
  const float* xr = xi + (size_t)b * Nf;
  float s = 0.f;
  for (int n = 0; n < Nf; ++n) s += ar[n] * xr[n];
  xc[idx] = s;
}

// s = softmax(assMatrix, axis=-1), rows of 50
__global__ void softmax_kernel(const float* __restrict__ aM, float* __restrict__ s) {
  int r = blockIdx.x * blockDim.x + threadIdx.x;
  if (r >= NCd) return;
  const float* row = aM + (size_t)r * NSd;
  float mx = row[0];
  for (int j = 1; j < NSd; ++j) mx = fmaxf(mx, row[j]);
  float sum = 0.f;
  for (int j = 0; j < NSd; ++j) sum += expf(row[j] - mx);
  float inv = 1.f / sum;
  for (int j = 0; j < NSd; ++j) s[(size_t)r * NSd + j] = expf(row[j] - mx) * inv;
}

// t = adj1_raw(500x500) @ s(500x50)
__global__ void t_kernel(const float* __restrict__ adj1, const float* __restrict__ s,
                         float* __restrict__ t) {
  int idx = blockIdx.x * blockDim.x + threadIdx.x;
  if (idx >= NCd * NSd) return;
  int j = idx % NSd, i = idx / NSd;
  float acc = 0.f;
  for (int k = 0; k < NCd; ++k) acc += adj1[(size_t)i * NCd + k] * s[(size_t)k * NSd + j];
  t[idx] = acc;
}

// adj2 = s^T(50x500) @ t(500x50)
__global__ void adj2_kernel(const float* __restrict__ s, const float* __restrict__ t,
                            float* __restrict__ adj2) {
  int idx = blockIdx.x * blockDim.x + threadIdx.x;
  if (idx >= NSd * NSd) return;
  int j = idx % NSd, i = idx / NSd;
  float acc = 0.f;
  for (int c = 0; c < NCd; ++c) acc += s[(size_t)c * NSd + i] * t[(size_t)c * NSd + j];
  adj2[idx] = acc;
}

// xs[b,sn] = sum_c s[c,sn] * xc[b,c]
__global__ void xs_kernel(const float* __restrict__ s, const float* __restrict__ xc,
                          float* __restrict__ xs) {
  int idx = blockIdx.x * blockDim.x + threadIdx.x;
  if (idx >= Bn * NSd) return;
  int sn = idx % NSd, b = idx / NSd;
  float acc = 0.f;
  for (int c = 0; c < NCd; ++c) acc += s[(size_t)c * NSd + sn] * xc[(size_t)b * NCd + c];
  xs[idx] = acc;
}

// ---------------------------------------------------------------------------

extern "C" void kernel_launch(void* const* d_in, const int* in_sizes, int n_in,
                              void* d_out, int out_size, void* d_ws, size_t ws_size,
                              hipStream_t stream) {
  const float* inputs = (const float*)d_in[0];
  const float* state  = (const float*)d_in[1];
  const float* state1 = (const float*)d_in[2];
  const float* state2 = (const float*)d_in[3];
  const float* A      = (const float*)d_in[4];
  const float* A1     = (const float*)d_in[5];
  const float* adj1   = (const float*)d_in[6];
  const float* afc    = (const float*)d_in[7];
  const float* assM   = (const float*)d_in[8];
  const float* W0  = (const float*)d_in[9];  const float* b0  = (const float*)d_in[10];
  const float* W1  = (const float*)d_in[11]; const float* b1  = (const float*)d_in[12];
  const float* W01 = (const float*)d_in[13]; const float* b01 = (const float*)d_in[14];
  const float* W11 = (const float*)d_in[15]; const float* b11 = (const float*)d_in[16];
  const float* W02 = (const float*)d_in[17]; const float* b02 = (const float*)d_in[18];
  const float* W12 = (const float*)d_in[19]; const float* b12 = (const float*)d_in[20];
  float* out = (float*)d_out;

  // workspace carve-out (256B aligned slabs)
  char* p = (char*)d_ws;
  auto take = [&](size_t bytes) -> char* {
    char* r = p; p += ((bytes + 255) & ~(size_t)255); return r;
  };
  __bf16* Abf   = (__bf16*)take((size_t)MPF * KPF * 2);
  __bf16* A1bf  = (__bf16*)take((size_t)MPC * KPC * 2);
  __bf16* A2bf  = (__bf16*)take((size_t)MPS * KPS * 2);
  float*  adj2f = (float*)take((size_t)NSd * NSd * 4);
  float*  sMat  = (float*)take((size_t)NCd * NSd * 4);
  float*  tMat  = (float*)take((size_t)NCd * NSd * 4);
  float*  xc    = (float*)take((size_t)Bn * NCd * 4);
  float*  xs    = (float*)take((size_t)Bn * NSd * 4);
  __bf16* XtF   = (__bf16*)take((size_t)CCOLP * KPF * 2);
  __bf16* XtC   = (__bf16*)take((size_t)CCOLP * KPC * 2);
  __bf16* XtS   = (__bf16*)take((size_t)CCOLP * KPS * 2);
  float*  YF    = (float*)take((size_t)MPF * CCOLP * 4);
  float*  YC    = (float*)take((size_t)MPC * CCOLP * 4);
  float*  YS    = (float*)take((size_t)MPS * CCOLP * 4);
  float*  uF    = (float*)take((size_t)Bn * Nf * Ud * 4);
  float*  uC    = (float*)take((size_t)Bn * NCd * Ud * 4);
  float*  uS    = (float*)take((size_t)Bn * NSd * Ud * 4);

  float* out0 = out;                                   // new_state  (B, N*U)
  float* out1 = out0 + (size_t)Bn * Nf * Ud;           // new_state1 (B, NC*U)
  float* out2 = out1 + (size_t)Bn * NCd * Ud;          // new_state2 (B, NS*U)

  auto nb = [](size_t n) { return (unsigned)((n + 255) / 256); };
  auto gemmBlocks = [](int Mp) { return (unsigned)(((size_t)(Mp / 16) * NGRP + 7) / 8); };

  // ---- graph prep ----
  pack_mat_bf16<<<nb((size_t)MPF * KPF), 256, 0, stream>>>(A,  Abf,  Nf,  Nf,  MPF, KPF);
  pack_mat_bf16<<<nb((size_t)MPC * KPC), 256, 0, stream>>>(A1, A1bf, NCd, NCd, MPC, KPC);
  xc_kernel<<<nb(Bn * NCd), 256, 0, stream>>>(afc, inputs, xc);
  softmax_kernel<<<nb(NCd), 256, 0, stream>>>(assM, sMat);
  t_kernel<<<nb(NCd * NSd), 256, 0, stream>>>(adj1, sMat, tMat);
  adj2_kernel<<<nb(NSd * NSd), 256, 0, stream>>>(sMat, tMat, adj2f);
  xs_kernel<<<nb(Bn * NSd), 256, 0, stream>>>(sMat, xc, xs);
  pack_mat_bf16<<<nb((size_t)MPS * KPS), 256, 0, stream>>>(adj2f, A2bf, NSd, NSd, MPS, KPS);

  // ---- fine level ----
  pack_x0<<<nb((size_t)CCOLP * KPF), 256, 0, stream>>>(XtF, inputs, state, Nf, KPF);
  wmma_gemm<<<gemmBlocks(MPF), 256, 0, stream>>>(Abf, XtF, YF, MPF, KPF);
  gate_kernel<<<nb((size_t)Bn * Nf * 128), 256, 0, stream>>>(YF, state, W0, b0, uF, XtF, Nf, KPF);
  wmma_gemm<<<gemmBlocks(MPF), 256, 0, stream>>>(Abf, XtF, YF, MPF, KPF);
  cand_kernel<<<nb((size_t)Bn * Nf * Ud), 256, 0, stream>>>(YF, state, W1, b1, uF, out0, Nf);

  // ---- coarse level ----
  pack_x0<<<nb((size_t)CCOLP * KPC), 256, 0, stream>>>(XtC, xc, state1, NCd, KPC);
  wmma_gemm<<<gemmBlocks(MPC), 256, 0, stream>>>(A1bf, XtC, YC, MPC, KPC);
  gate_kernel<<<nb((size_t)Bn * NCd * 128), 256, 0, stream>>>(YC, state1, W01, b01, uC, XtC, NCd, KPC);
  wmma_gemm<<<gemmBlocks(MPC), 256, 0, stream>>>(A1bf, XtC, YC, MPC, KPC);
  cand_kernel<<<nb((size_t)Bn * NCd * Ud), 256, 0, stream>>>(YC, state1, W11, b11, uC, out1, NCd);

  // ---- super level ----
  pack_x0<<<nb((size_t)CCOLP * KPS), 256, 0, stream>>>(XtS, xs, state2, NSd, KPS);
  wmma_gemm<<<gemmBlocks(MPS), 256, 0, stream>>>(A2bf, XtS, YS, MPS, KPS);
  gate_kernel<<<nb((size_t)Bn * NSd * 128), 256, 0, stream>>>(YS, state2, W02, b02, uS, XtS, NSd, KPS);
  wmma_gemm<<<gemmBlocks(MPS), 256, 0, stream>>>(A2bf, XtS, YS, MPS, KPS);
  cand_kernel<<<nb((size_t)Bn * NSd * Ud), 256, 0, stream>>>(YS, state2, W12, b12, uS, out2, NSd);
}
